// SpGAT_34488587387254
// MI455X (gfx1250) — matrix operand order repaired and checked by reading, compiled, not run on hardware
//
#include <hip/hip_runtime.h>
#include <stdint.h>

// ---------------------------------------------------------------------------
// SpGAT (2-layer sparse graph attention) for MI455X / gfx1250, wave32 + WMMA.
//   Layer1: h1 = x @ W1cat (bf16 WMMA, f32 accum) -> per-head scores ->
//           per-edge softmax-weights -> atomic scatter agg -> elu -> hcat(bf16)
//   Layer2: h2 = hcat @ W2pad -> scores -> edge scatter -> elu -> out
// A-tiles staged via async global->LDS (ASYNCcnt); edge gathers prefetched.
// ---------------------------------------------------------------------------

#define N_NODES 50000
#define IN_F    512
#define HID     64
#define HEADS   8
#define N_CLS   20
#define N_CLSP  32   // padded to WMMA tile multiple

typedef __attribute__((ext_vector_type(16))) __bf16 v16bf;
typedef __attribute__((ext_vector_type(8)))  float  v8f;

union BF16Frag { uint32_t u[8]; v16bf v; };

__device__ __forceinline__ unsigned short f2bf(float f) {
  unsigned int x = __float_as_uint(f);
  unsigned int r = x + 0x7FFFu + ((x >> 16) & 1u);   // round-to-nearest-even
  return (unsigned short)(r >> 16);
}

__device__ __forceinline__ uint32_t pack_bf16(float a, float b) {
  return (uint32_t)f2bf(a) | ((uint32_t)f2bf(b) << 16);
}

__device__ __forceinline__ float elu1(float v) {
  return v > 0.f ? v : (__expf(v) - 1.f);
}

// CDNA5 async global->LDS copy (16B per lane), tracked on ASYNCcnt.
// LDS offset = low 32 bits of the flat shared-pointer (aperture has low32=0).
__device__ __forceinline__ void async_ld_b128(uint32_t lds_off,
                                              unsigned long long gaddr) {
  asm volatile("global_load_async_to_lds_b128 %0, %1, off"
               :: "v"(lds_off), "v"(gaddr) : "memory");
}
__device__ __forceinline__ void wait_async0() {
  asm volatile("s_wait_asynccnt 0x0" ::: "memory");
}

// ---------------------------------------------------------------------------
// f32 -> bf16 bulk convert (4 elems / thread)
// ---------------------------------------------------------------------------
__global__ void f32_to_bf16_kernel(const float* __restrict__ src,
                                   unsigned short* __restrict__ dst, int nElem) {
  int i = (blockIdx.x * 256 + threadIdx.x) * 4;
  if (i >= nElem) return;
  float4 v = *reinterpret_cast<const float4*>(src + i);
  uint2 p;
  p.x = pack_bf16(v.x, v.y);
  p.y = pack_bf16(v.z, v.w);
  *reinterpret_cast<uint2*>(dst + i) = p;
}

// Weight repack: W1 [8][512][64] f32 -> Bbf1 [512][512] bf16 (heads on N)
__global__ void repack_w1_kernel(const float* __restrict__ W1,
                                 unsigned short* __restrict__ Bbf) {
  int t = blockIdx.x * 256 + threadIdx.x;
  if (t >= IN_F * IN_F) return;
  int k = t >> 9, c = t & 511;
  int h = c >> 6, n = c & 63;
  Bbf[t] = f2bf(W1[(size_t)h * IN_F * HID + (size_t)k * HID + n]);
}

// W2 [512][20] f32 -> Bbf2 [512][32] bf16 zero-padded
__global__ void repack_w2_kernel(const float* __restrict__ W2,
                                 unsigned short* __restrict__ Bbf) {
  int t = blockIdx.x * 256 + threadIdx.x;
  if (t >= IN_F * N_CLSP) return;
  int k = t >> 5, c = t & 31;
  Bbf[t] = (c < N_CLS) ? f2bf(W2[(size_t)k * N_CLS + c]) : (unsigned short)0;
}

// ---------------------------------------------------------------------------
// bf16 WMMA GEMM: C[M,Ncols] = A[M,K](bf16) @ B[K,Ncols](bf16)
// 256 threads = 8 waves; tile 128(M) x 64(N) x 32(K); wave strip = 16x64.
// A tile: async global->LDS b128 copies. B tile: sync transposed staging.
// ---------------------------------------------------------------------------
__global__ __launch_bounds__(256) void gemm_bf16_wmma_kernel(
    const unsigned short* __restrict__ A, const unsigned short* __restrict__ B,
    float* __restrict__ C, int M, int K, int Ncols) {
  // Row pitch 40 ushorts (80B): conflict-free across 16 rows, keeps 16B align.
  __shared__ __align__(16) unsigned short As[128 * 40];
  __shared__ __align__(16) unsigned short Bs[64 * 40];

  const int tid    = threadIdx.x;
  const int lane   = tid & 31;
  const int wv     = tid >> 5;       // wave 0..7 -> M rows wv*16..+15
  const int half   = lane >> 4;      // 0 | 1
  const int m16    = lane & 15;
  const int blockN = blockIdx.x * 64;
  const int blockM = blockIdx.y * 128;

  v8f acc0 = {}, acc1 = {}, acc2 = {}, acc3 = {};

  for (int kb = 0; kb < K; kb += 32) {
    __syncthreads();
    // ---- A tile: 128x32 bf16 = 512 x 16B chunks, async copied to LDS ----
#pragma unroll
    for (int i = 0; i < 2; ++i) {
      int c  = tid + i * 256;           // chunk 0..511
      int r  = c >> 2;                  // row in tile
      int c8 = (c & 3) << 3;            // ushort col (0,8,16,24)
      int gm = blockM + r;
      if (gm >= M) gm = M - 1;          // clamp: valid addr, no divergence
      async_ld_b128((uint32_t)(uintptr_t)&As[r * 40 + c8],
                    (unsigned long long)(A + (size_t)gm * K + kb + c8));
    }
    // ---- B tile transposed: Bs[n][k], coalesced u16 global reads ----
#pragma unroll
    for (int i = 0; i < 8; ++i) {
      int idx = tid + i * 256;          // 0..2047
      int k   = idx >> 6;               // 0..31
      int n   = idx & 63;
      int gn  = blockN + n;
      if (gn >= Ncols) gn = Ncols - 1;  // clamp feeds only unstored acc2/3
      Bs[n * 40 + k] = B[(size_t)(kb + k) * Ncols + gn];
    }
    wait_async0();
    __syncthreads();

    // ---- A fragment (row m16, contiguous k = half*16 + s, as B) ----
    BF16Frag fa;
    const int rowA = (wv << 4) + m16;
    *reinterpret_cast<uint4*>(&fa.u[0]) =
        *reinterpret_cast<const uint4*>(&As[rowA * 40 + half * 16]);
    *reinterpret_cast<uint4*>(&fa.u[4]) =
        *reinterpret_cast<const uint4*>(&As[rowA * 40 + half * 16 + 8]);

    // ---- all four B fragments up front: batch ds_loads, then 4x WMMA ----
    BF16Frag fb0, fb1, fb2, fb3;
    {
      const unsigned short* bp = &Bs[(size_t)m16 * 40 + half * 16];
      *reinterpret_cast<uint4*>(&fb0.u[0]) = *reinterpret_cast<const uint4*>(bp + 0 * 16 * 40);
      *reinterpret_cast<uint4*>(&fb0.u[4]) = *reinterpret_cast<const uint4*>(bp + 0 * 16 * 40 + 8);
      *reinterpret_cast<uint4*>(&fb1.u[0]) = *reinterpret_cast<const uint4*>(bp + 1 * 16 * 40);
      *reinterpret_cast<uint4*>(&fb1.u[4]) = *reinterpret_cast<const uint4*>(bp + 1 * 16 * 40 + 8);
      *reinterpret_cast<uint4*>(&fb2.u[0]) = *reinterpret_cast<const uint4*>(bp + 2 * 16 * 40);
      *reinterpret_cast<uint4*>(&fb2.u[4]) = *reinterpret_cast<const uint4*>(bp + 2 * 16 * 40 + 8);
      *reinterpret_cast<uint4*>(&fb3.u[0]) = *reinterpret_cast<const uint4*>(bp + 3 * 16 * 40);
      *reinterpret_cast<uint4*>(&fb3.u[4]) = *reinterpret_cast<const uint4*>(bp + 3 * 16 * 40 + 8);
    }
    acc0 = __builtin_amdgcn_wmma_f32_16x16x32_bf16(false, fa.v, false, fb0.v,
                                                   (short)0, acc0, false, false);
    acc1 = __builtin_amdgcn_wmma_f32_16x16x32_bf16(false, fa.v, false, fb1.v,
                                                   (short)0, acc1, false, false);
    acc2 = __builtin_amdgcn_wmma_f32_16x16x32_bf16(false, fa.v, false, fb2.v,
                                                   (short)0, acc2, false, false);
    acc3 = __builtin_amdgcn_wmma_f32_16x16x32_bf16(false, fa.v, false, fb3.v,
                                                   (short)0, acc3, false, false);
  }

  // ---- store C (D layout: VGPR r -> M = half*8+r, N = lane%16) ----
  const int gm0 = blockM + (wv << 4) + (half << 3);
  float* Cp = C + (size_t)gm0 * Ncols + blockN + m16;
  if (blockM + 128 <= M && blockN + 64 <= Ncols) {
    // uniform fast path: no guards, constant offsets from one base
#pragma unroll
    for (int r = 0; r < 8; ++r) {
      float* p = Cp + (size_t)r * Ncols;
      p[0]  = acc0[r];
      p[16] = acc1[r];
      p[32] = acc2[r];
      p[48] = acc3[r];
    }
  } else {
#pragma unroll
    for (int r = 0; r < 8; ++r) {
      int gm = gm0 + r;
      if (gm < M) {
        int gn = blockN + m16;
        float* p = Cp + (size_t)r * Ncols;
        if (gn      < Ncols) p[0]  = acc0[r];
        if (gn + 16 < Ncols) p[16] = acc1[r];
        if (gn + 32 < Ncols) p[32] = acc2[r];
        if (gn + 48 < Ncols) p[48] = acc3[r];
      }
    }
  }
}

// ---------------------------------------------------------------------------
// Layer-1 attention scores: wave per node; lane l -> head l/4, quarter l%4.
// ---------------------------------------------------------------------------
__global__ __launch_bounds__(256) void scores1_kernel(
    const float* __restrict__ h1, const float* __restrict__ a1,
    float* __restrict__ ssrc, float* __restrict__ sdst, int n) {
  int node = blockIdx.x * 8 + (threadIdx.x >> 5);
  if (node >= n) return;                       // wave-uniform exit
  int lane = threadIdx.x & 31;
  int h = lane >> 2, q = lane & 3;
  const float* row = h1 + (size_t)node * IN_F + (lane << 4);
  const float* as  = a1 + h * (2 * HID) + q * 16;
  const float* ad  = as + HID;
  float s0 = 0.f, s1 = 0.f;
#pragma unroll
  for (int j = 0; j < 16; ++j) {
    float v = row[j];
    s0 = fmaf(v, as[j], s0);
    s1 = fmaf(v, ad[j], s1);
  }
  s0 += __shfl_xor(s0, 1, 32); s0 += __shfl_xor(s0, 2, 32);
  s1 += __shfl_xor(s1, 1, 32); s1 += __shfl_xor(s1, 2, 32);
  if (q == 0) {
    ssrc[h * n + node] = s0;
    sdst[h * n + node] = s1;
  }
}

// ---------------------------------------------------------------------------
// Layer-1 edge scatter: wave32 per 4 edges; lane owns 16 features (head l/4).
// Gathers (h1) and scatters (agg) both L2-resident (102MB each vs 192MB L2).
// Prefetch all 4 gather rows, then non-returning f32 atomics.
// ---------------------------------------------------------------------------
__global__ __launch_bounds__(256) void edge_agg1_kernel(
    const int* __restrict__ e0, const int* __restrict__ e1,
    const float* __restrict__ ssrc, const float* __restrict__ sdst,
    const float* __restrict__ h1, float* __restrict__ rowsum,
    float* __restrict__ agg, int E, int n) {
  int wid  = (blockIdx.x * 256 + threadIdx.x) >> 5;
  int base = wid * 4;
  if (base >= E) return;                       // wave-uniform exit
  int lane = threadIdx.x & 31;
  int h = lane >> 2, q = lane & 3;
  int nE = E - base; if (nE > 4) nE = 4;
  int us[4], vs[4];
#pragma unroll
  for (int e = 0; e < 4; ++e) {
    int idx = base + ((e < nE) ? e : 0);
    us[e] = e0[idx];
    vs[e] = e1[idx];
    __builtin_prefetch(h1 + (size_t)vs[e] * IN_F + (lane << 4), 0, 0);
  }
  for (int e = 0; e < nE; ++e) {
    int u = us[e], v = vs[e];
    float s = ssrc[h * n + u] + sdst[h * n + v];
    s = (s > 0.f) ? s : 0.2f * s;              // LeakyReLU(0.2)
    float w = __expf(s);
    if (q == 0) unsafeAtomicAdd(&rowsum[h * n + u], w);
    const float* hp = h1  + (size_t)v * IN_F + (lane << 4);
    float*       op = agg + (size_t)u * IN_F + (lane << 4);
#pragma unroll
    for (int j = 0; j < 4; ++j) {
      float4 x = *reinterpret_cast<const float4*>(hp + j * 4);
      unsafeAtomicAdd(op + j * 4 + 0, w * x.x);
      unsafeAtomicAdd(op + j * 4 + 1, w * x.y);
      unsafeAtomicAdd(op + j * 4 + 2, w * x.z);
      unsafeAtomicAdd(op + j * 4 + 3, w * x.w);
    }
  }
}

// hcat = elu(agg / rowsum) -> bf16 (2 elems/thread; pair stays in one head)
__global__ void hcat_elu_bf16_kernel(const float* __restrict__ agg,
                                     const float* __restrict__ rowsum,
                                     unsigned short* __restrict__ dst, int n) {
  int i2 = (blockIdx.x * 256 + threadIdx.x) * 2;
  if (i2 >= n * IN_F) return;
  int node = i2 >> 9;
  int hh   = (i2 & 511) >> 6;
  float r  = rowsum[hh * n + node];
  float v0 = elu1(agg[i2]     / r);
  float v1 = elu1(agg[i2 + 1] / r);
  *reinterpret_cast<uint32_t*>(dst + i2) = pack_bf16(v0, v1);
}

// Layer-2 scores (single "head", 20 features; h2 pitch = 32)
__global__ void scores2_kernel(const float* __restrict__ h2,
                               const float* __restrict__ a2,
                               float* __restrict__ ss, float* __restrict__ sd,
                               int n) {
  int i = blockIdx.x * 256 + threadIdx.x;
  if (i >= n) return;
  const float* r = h2 + (size_t)i * N_CLSP;
  float s0 = 0.f, s1 = 0.f;
#pragma unroll
  for (int c = 0; c < N_CLS; ++c) {
    float v = r[c];
    s0 = fmaf(v, a2[c], s0);
    s1 = fmaf(v, a2[N_CLS + c], s1);
  }
  ss[i] = s0; sd[i] = s1;
}

__global__ __launch_bounds__(256) void edge_agg2_kernel(
    const int* __restrict__ e0, const int* __restrict__ e1,
    const float* __restrict__ ss, const float* __restrict__ sd,
    const float* __restrict__ h2, float* __restrict__ rowsum,
    float* __restrict__ out, int E) {
  int wid = (blockIdx.x * 256 + threadIdx.x) >> 5;
  if (wid >= E) return;
  int lane = threadIdx.x & 31;
  int u = e0[wid], v = e1[wid];
  float s = ss[u] + sd[v];
  s = (s > 0.f) ? s : 0.2f * s;
  float w = __expf(s);
  if (lane == 0) unsafeAtomicAdd(&rowsum[u], w);
  if (lane < N_CLS)
    unsafeAtomicAdd(out + (size_t)u * N_CLS + lane,
                    w * h2[(size_t)v * N_CLSP + lane]);
}

__global__ void finalize_kernel(float* __restrict__ out,
                                const float* __restrict__ rowsum, int n) {
  int t = blockIdx.x * 256 + threadIdx.x;
  if (t >= n * N_CLS) return;
  out[t] = elu1(out[t] / rowsum[t / N_CLS]);
}

// ---------------------------------------------------------------------------
extern "C" void kernel_launch(void* const* d_in, const int* in_sizes, int n_in,
                              void* d_out, int out_size, void* d_ws, size_t ws_size,
                              hipStream_t stream) {
  const float* x  = (const float*)d_in[0];
  const float* W1 = (const float*)d_in[1];
  const float* a1 = (const float*)d_in[2];
  const float* W2 = (const float*)d_in[3];
  const float* a2 = (const float*)d_in[4];
  const int*   ei = (const int*)d_in[5];
  const int n = in_sizes[0] / IN_F;           // 50000
  const int E = in_sizes[5] / 2;              // 1,650,000
  float* out = (float*)d_out;

  // ---- workspace carve (256B aligned) ----
  char* w = (char*)d_ws;
  size_t off = 0;
  auto carve = [&](size_t bytes) {
    void* p = w + off;
    off += (bytes + 255) & ~(size_t)255;
    return p;
  };
  float*          h1      = (float*)carve((size_t)n * IN_F * 4);   // 102.4 MB
  float*          agg1    = (float*)carve((size_t)n * IN_F * 4);   // 102.4 MB
  unsigned short* abf     = (unsigned short*)carve((size_t)n * IN_F * 2); // x/hcat bf16
  unsigned short* Bbf1    = (unsigned short*)carve((size_t)IN_F * IN_F * 2);
  unsigned short* Bbf2    = (unsigned short*)carve((size_t)IN_F * N_CLSP * 2 + 256);
  float*          ssrc1   = (float*)carve((size_t)HEADS * n * 4);
  float*          sdst1   = (float*)carve((size_t)HEADS * n * 4);
  float*          rowsum1 = (float*)carve((size_t)HEADS * n * 4);
  float*          h2      = (float*)carve((size_t)n * N_CLSP * 4);
  float*          ssrc2   = (float*)carve((size_t)n * 4);
  float*          sdst2   = (float*)carve((size_t)n * 4);
  float*          rowsum2 = (float*)carve((size_t)n * 4);
  (void)ws_size; (void)n_in; (void)out_size;

  // ---- zero accumulators (every call: graph-replay deterministic) ----
  hipMemsetAsync(agg1,    0, (size_t)n * IN_F * 4, stream);
  hipMemsetAsync(rowsum1, 0, (size_t)HEADS * n * 4, stream);
  hipMemsetAsync(rowsum2, 0, (size_t)n * 4, stream);
  hipMemsetAsync(out,     0, (size_t)n * N_CLS * 4, stream);

  // ---- precision prep: x->bf16, weights->bf16 ----
  f32_to_bf16_kernel<<<(n * IN_F / 4 + 255) / 256, 256, 0, stream>>>(
      x, abf, n * IN_F);
  repack_w1_kernel<<<(IN_F * IN_F + 255) / 256, 256, 0, stream>>>(W1, Bbf1);
  repack_w2_kernel<<<(IN_F * N_CLSP + 255) / 256, 256, 0, stream>>>(W2, Bbf2);

  // ---- layer 1 ----
  {
    dim3 grid(IN_F / 64, (n + 127) / 128);          // (8, 391)
    gemm_bf16_wmma_kernel<<<grid, 256, 0, stream>>>(abf, Bbf1, h1, n, IN_F, IN_F);
  }
  scores1_kernel<<<(n + 7) / 8, 256, 0, stream>>>(h1, a1, ssrc1, sdst1, n);
  {
    long long waves = ((long long)E + 3) / 4;
    int blocks = (int)((waves * 32 + 255) / 256);
    edge_agg1_kernel<<<blocks, 256, 0, stream>>>(ei, ei + E, ssrc1, sdst1,
                                                 h1, rowsum1, agg1, E, n);
  }
  hcat_elu_bf16_kernel<<<(n * IN_F / 2 + 255) / 256, 256, 0, stream>>>(
      agg1, rowsum1, abf, n);

  // ---- layer 2 ----
  {
    dim3 grid(1, (n + 127) / 128);
    gemm_bf16_wmma_kernel<<<grid, 256, 0, stream>>>(abf, Bbf2, h2, n, IN_F, N_CLSP);
  }
  scores2_kernel<<<(n + 255) / 256, 256, 0, stream>>>(h2, a2, ssrc2, sdst2, n);
  {
    int blocks = (int)(((long long)E * 32 + 255) / 256);
    edge_agg2_kernel<<<blocks, 256, 0, stream>>>(ei, ei + E, ssrc2, sdst2,
                                                 h2, rowsum2, out, E);
  }
  finalize_kernel<<<(n * N_CLS + 255) / 256, 256, 0, stream>>>(out, rowsum2, n);
}
